// S4_38414187495633
// MI455X (gfx1250) — matrix operand order repaired and checked by reading, not compiled, not run on hardware
//
#include <hip/hip_runtime.h>
#include <math.h>

// CDNA5 / gfx1250: wave32, WMMA 16x16x32 f16 -> f32 accumulate.
typedef __attribute__((ext_vector_type(16))) _Float16 v16h;
typedef __attribute__((ext_vector_type(8)))  _Float16 v8h;
typedef __attribute__((ext_vector_type(8)))  float    v8f;

#define S4_B    32
#define S4_L    4096
#define S4_DIN  32
#define S4_H    256
#define S4_N    32

// ---------------------------------------------------------------------------
// Kernel 0: zero accumulators (acc[B*H] and kksum[H] are contiguous in ws)
// ---------------------------------------------------------------------------
__global__ __launch_bounds__(256) void s4_zero(float* __restrict__ p, int n)
{
    int i = blockIdx.x * 256 + threadIdx.x;
    if (i < n) p[i] = 0.f;
}

// ---------------------------------------------------------------------------
// Kernel 1: per-(h,n) SSM coefficients:  dtA = dt*A,  Kc = C*(exp(dtA)-1)/A
// ---------------------------------------------------------------------------
__global__ __launch_bounds__(256) void s4_coef(
    const float* __restrict__ log_dt, const float* __restrict__ A_re,
    const float* __restrict__ A_im,   const float* __restrict__ C_re,
    const float* __restrict__ C_im,
    float* __restrict__ car, float* __restrict__ cai,
    float* __restrict__ ckr, float* __restrict__ cki)
{
    int i = blockIdx.x * 256 + threadIdx.x;
    if (i >= S4_H * S4_N) return;
    int h = i / S4_N;
    float dt  = expf(log_dt[h]);
    float are = A_re[i], aim = A_im[i];
    float ar  = dt * are, ai = dt * aim;
    float er  = expf(ar), s, c;
    sincosf(ai, &s, &c);
    float wre = er * c - 1.0f;                   // exp(dtA) - 1
    float wim = er * s;
    float inv = 1.0f / (are * are + aim * aim);
    float dre = (wre * are + wim * aim) * inv;   // (exp(dtA)-1)/A
    float dim = (wim * are - wre * aim) * inv;
    float cr = C_re[i], ci = C_im[i];
    ckr[i] = cr * dre - ci * dim;                // Kc = C * ((exp(dtA)-1)/A)
    cki[i] = cr * dim + ci * dre;
    car[i] = ar;
    cai[i] = ai;
}

// ---------------------------------------------------------------------------
// Kernel 2: kk[h,j] = 2*Re( sum_n Kc[h,n] * exp(dtA[h,n]*(L-1-j)) )
//                    + D[h]*[j==L-1];  also accumulate kksum[h] = sum_j kk[h,j].
// Complex-geometric recurrence per 32-wide j-chunk.
// ---------------------------------------------------------------------------
__global__ __launch_bounds__(256) void s4_kk(
    const float* __restrict__ car, const float* __restrict__ cai,
    const float* __restrict__ ckr, const float* __restrict__ cki,
    const float* __restrict__ Dvec, float* __restrict__ kk,
    float* __restrict__ kksum)
{
    const int JC = S4_L / 32;
    int idx = blockIdx.x * 256 + threadIdx.x;
    if (idx >= S4_H * JC) return;
    int h  = idx / JC;
    int j0 = (idx % JC) * 32;

    float accv[32];
#pragma unroll
    for (int j = 0; j < 32; ++j) accv[j] = 0.f;

    float m0 = (float)(S4_L - 1 - j0);
    for (int n = 0; n < S4_N; ++n) {
        int hn = h * S4_N + n;
        float ar = car[hn], ai = cai[hn];
        float kr = ckr[hn], ki = cki[hn];
        float er = expf(ar * m0), s, c;
        sincosf(ai * m0, &s, &c);
        float cre = er * c, cim = er * s;           // exp(dtA*m0)
        float es = expf(-ar), ss, cs;
        sincosf(ai, &ss, &cs);
        float str = es * cs, sti = -es * ss;        // exp(-dtA)
#pragma unroll
        for (int j = 0; j < 32; ++j) {
            accv[j] += kr * cre - ki * cim;         // Re(Kc * cur)
            float nr = cre * str - cim * sti;
            cim      = cre * sti + cim * str;
            cre      = nr;
        }
    }
    float dh = Dvec[h];
    float csum = 0.f;
#pragma unroll
    for (int j = 0; j < 32; ++j) {
        int jj = j0 + j;
        float v = 2.f * accv[j];
        if (jj == S4_L - 1) v += dh;                // fold D*u term into kk
        csum += v;
        kk[(size_t)h * S4_L + jj] = v;
    }
    atomicAdd(&kksum[h], csum);                     // for b2-bias folding
}

// ---------------------------------------------------------------------------
// Kernel 3: pre-swizzle w1 (32x128) and w2 (128x256) into f16 WMMA B-fragments.
// B-fragment (KxN tile, K=32, N=16), lane ln: n = ln&15, kh = ln>>4,
// element e (0..15) holds B[e<8 ? kh*8+e : 16+kh*8+(e-8), n].
// ---------------------------------------------------------------------------
__global__ __launch_bounds__(256) void s4_wfrag(
    const float* __restrict__ w1, const float* __restrict__ w2,
    _Float16* __restrict__ w1f, _Float16* __restrict__ w2f)
{
    const int N1 = 8 * 32 * 16;      // 8 tiles for layer 1
    const int N2 = 64 * 32 * 16;     // 4 K-chunks * 16 tiles for layer 2
    int idx = blockIdx.x * 256 + threadIdx.x;
    if (idx < N1) {
        int t  = idx >> 9;
        int ln = (idx >> 4) & 31;
        int e  = idx & 15;
        int n  = ln & 15, kh = ln >> 4;
        int kl = (e < 8) ? kh * 8 + e : 16 + kh * 8 + (e - 8);
        w1f[idx] = (_Float16)w1[kl * 128 + t * 16 + n];
    } else if (idx < N1 + N2) {
        int j  = idx - N1;
        int f  = j >> 9;             // 0..63
        int kc = f >> 4, t = f & 15;
        int ln = (j >> 4) & 31;
        int e  = j & 15;
        int n  = ln & 15, kh = ln >> 4;
        int kl = (e < 8) ? kh * 8 + e : 16 + kh * 8 + (e - 8);
        w2f[j] = (_Float16)w2[(kc * 32 + kl) * 256 + t * 16 + n];
    }
}

// ---------------------------------------------------------------------------
// Kernel 4: convert data (B*L*32 f32) to f16 once; main kernel then loads
// A-fragments as aligned 16B vectors with zero conversion VALU in hot loop.
// ---------------------------------------------------------------------------
__global__ __launch_bounds__(256) void s4_cvt(
    const float* __restrict__ src, _Float16* __restrict__ dst)
{
    int i = (blockIdx.x * 256 + threadIdx.x) * 8;
#pragma unroll
    for (int e = 0; e < 8; ++e) dst[i + e] = (_Float16)src[i + e];
}

// ---------------------------------------------------------------------------
// Kernel 5 (main, WMMA): per block = one (b, 128-row l-chunk); per wave =
// one 16-row slab.  Layer1: 8 WMMAs K=32; relu -> LDS relayout; Layer2:
// 16 tiles x 4 K-chunks WMMAs; then kk-weighted reduction over l into acc[b,h].
// ---------------------------------------------------------------------------
__global__ __launch_bounds__(256) void s4_main(
    const _Float16* __restrict__ d16, const float* __restrict__ b1,
    const _Float16* __restrict__ w1f, const _Float16* __restrict__ w2f,
    const float* __restrict__ kk, float* __restrict__ acc)
{
    __shared__ _Float16 stage[8][16][136];   // per-wave 16x128 X1 tile (+pad)
    __shared__ float lacc[S4_H];

    int tid  = threadIdx.x;
    int ln   = tid & 31, wave = tid >> 5;
    int b    = blockIdx.x >> 5;
    int chnk = blockIdx.x & 31;
    int lbase = chnk * 128 + wave * 16;

    lacc[tid] = 0.f;
    __syncthreads();

    int m = ln & 15, kh = ln >> 4;

    // warm L2/WGP$ for the kk rows this lane will reduce with at the end
#pragma unroll
    for (int t = 0; t < 16; ++t)
        __builtin_prefetch(kk + (size_t)(t * 16 + m) * S4_L + lbase + 8 * kh, 0, 1);

    // A-fragment of data rows (16x32 f16): lane holds row m, K split 8/8 per half
    const _Float16* drow = d16 + ((size_t)b * S4_L + (lbase + m)) * S4_DIN;
    v8h lo0 = *(const v8h*)(drow + kh * 8);
    v8h hi0 = *(const v8h*)(drow + 16 + kh * 8);
    v16h a0 = __builtin_shufflevector(lo0, hi0, 0, 1, 2, 3, 4, 5, 6, 7,
                                      8, 9, 10, 11, 12, 13, 14, 15);

    // ---- layer 1: x1 = relu(data@w1 + b1), 8 output tiles of 16 ----
#pragma unroll
    for (int t = 0; t < 8; ++t) {
        v16h bf = *(const v16h*)(w1f + ((t * 32 + ln) << 4));
        v8f  c  = {};
        c = __builtin_amdgcn_wmma_f32_16x16x32_f16(false, a0, false, bf,
                                                   (short)0, c, false, false);
        int col = t * 16 + m;                 // C-layout: lane -> column n
        float bias = b1[col];
#pragma unroll
        for (int v = 0; v < 8; ++v) {
            float x = c[v] + bias;
            stage[wave][v + 8 * kh][col] = (_Float16)fmaxf(x, 0.f);
        }
    }
    __syncthreads();

    // reload as A-fragments (16x32 per K-chunk), two aligned b128 LDS loads each
    v16h a1[4];
#pragma unroll
    for (int kc = 0; kc < 4; ++kc) {
        v8h lo = *(const v8h*)(&stage[wave][m][kc * 32 + kh * 8]);
        v8h hi = *(const v8h*)(&stage[wave][m][kc * 32 + 16 + kh * 8]);
        a1[kc] = __builtin_shufflevector(lo, hi, 0, 1, 2, 3, 4, 5, 6, 7,
                                         8, 9, 10, 11, 12, 13, 14, 15);
    }

    // ---- layer 2 (bias folded out via kksum) + kk-weighted reduction ----
    for (int t = 0; t < 16; ++t) {
        v8f c = {};
#pragma unroll
        for (int kc = 0; kc < 4; ++kc) {
            v16h bf = *(const v16h*)(w2f + (((kc * 16 + t) * 32 + ln) << 4));
            c = __builtin_amdgcn_wmma_f32_16x16x32_f16(false, a1[kc], false, bf,
                                                       (short)0, c, false, false);
        }
        int h = t * 16 + m;
        const float* kr = kk + (size_t)h * S4_L + lbase + 8 * kh;  // contiguous in l
        float p = 0.f;
#pragma unroll
        for (int v = 0; v < 8; ++v) p += c[v] * kr[v];
        atomicAdd(&lacc[h], p);               // ds_add_f32
    }
    __syncthreads();
    atomicAdd(&acc[b * S4_H + tid], lacc[tid]);  // global_atomic_add_f32
}

// ---------------------------------------------------------------------------
// Kernel 6 (tail): +b2*kksum -> gelu -> @Wg + gate -> relu(@w3) -> @w4.
// One block per b.
// ---------------------------------------------------------------------------
__global__ __launch_bounds__(256) void s4_tail(
    const float* __restrict__ acc, const float* __restrict__ b2,
    const float* __restrict__ kksum,
    const float* __restrict__ Wg,  const float* __restrict__ bg,
    const float* __restrict__ w3,  const float* __restrict__ b3,
    const float* __restrict__ w4,  const float* __restrict__ b4,
    float* __restrict__ out)
{
    __shared__ float yg[S4_H];
    __shared__ float y2[S4_H];
    __shared__ float zz[128];
    int b = blockIdx.x, t = threadIdx.x;

    float x = acc[b * S4_H + t] + b2[t] * kksum[t];
    yg[t] = 0.5f * x * (1.f + erff(x * 0.7071067811865475f));   // exact gelu
    __syncthreads();

    float va = bg[t], vb = bg[t + S4_H];
    for (int h = 0; h < S4_H; ++h) {
        float yh = yg[h];
        va += yh * Wg[h * 2 * S4_H + t];
        vb += yh * Wg[h * 2 * S4_H + t + S4_H];
    }
    y2[t] = va * (1.f / (1.f + expf(-vb)));
    __syncthreads();

    if (t < 128) {
        float s = b3[t];
        for (int h = 0; h < S4_H; ++h) s += y2[h] * w3[h * 128 + t];
        zz[t] = fmaxf(s, 0.f);
    }
    __syncthreads();

    if (t < 16) {
        float s = b4[t];
        for (int j = 0; j < 128; ++j) s += zz[j] * w4[j * 16 + t];
        out[b * 16 + t] = s;
    }
}

// ---------------------------------------------------------------------------
extern "C" void kernel_launch(void* const* d_in, const int* in_sizes, int n_in,
                              void* d_out, int out_size, void* d_ws, size_t ws_size,
                              hipStream_t stream) {
    const float* data  = (const float*)d_in[0];
    const float* w1    = (const float*)d_in[1];
    const float* b1    = (const float*)d_in[2];
    const float* w2    = (const float*)d_in[3];
    const float* b2    = (const float*)d_in[4];
    const float* logdt = (const float*)d_in[5];
    const float* A_re  = (const float*)d_in[6];
    const float* A_im  = (const float*)d_in[7];
    const float* C_re  = (const float*)d_in[8];
    const float* C_im  = (const float*)d_in[9];
    const float* Dv    = (const float*)d_in[10];
    const float* Wg    = (const float*)d_in[11];
    const float* bg    = (const float*)d_in[12];
    const float* w3    = (const float*)d_in[13];
    const float* b3    = (const float*)d_in[14];
    const float* w4    = (const float*)d_in[15];
    const float* b4    = (const float*)d_in[16];
    float* out = (float*)d_out;

    // workspace carve (~12.8 MB total)
    char* p = (char*)d_ws;
    float* car  = (float*)p; p += (size_t)S4_H * S4_N * sizeof(float);
    float* cai  = (float*)p; p += (size_t)S4_H * S4_N * sizeof(float);
    float* ckr  = (float*)p; p += (size_t)S4_H * S4_N * sizeof(float);
    float* cki  = (float*)p; p += (size_t)S4_H * S4_N * sizeof(float);
    float* kkb  = (float*)p; p += (size_t)S4_H * S4_L * sizeof(float);
    float* accb = (float*)p; p += (size_t)S4_B * S4_H * sizeof(float);
    float* kks  = (float*)p; p += (size_t)S4_H * sizeof(float);          // contiguous with accb
    _Float16* w1f = (_Float16*)p; p += (size_t)8  * 32 * 16 * sizeof(_Float16);
    _Float16* w2f = (_Float16*)p; p += (size_t)64 * 32 * 16 * sizeof(_Float16);
    _Float16* d16 = (_Float16*)p; p += (size_t)S4_B * S4_L * S4_DIN * sizeof(_Float16);
    (void)ws_size; (void)n_in; (void)in_sizes; (void)out_size;

    const int nzero = S4_B * S4_H + S4_H;   // accb + kks (contiguous)
    s4_zero <<<(nzero + 255) / 256, 256, 0, stream>>>(accb, nzero);
    s4_coef <<<(S4_H * S4_N + 255) / 256, 256, 0, stream>>>(
        logdt, A_re, A_im, C_re, C_im, car, cai, ckr, cki);
    s4_wfrag<<<(8 * 32 * 16 + 64 * 32 * 16 + 255) / 256, 256, 0, stream>>>(
        w1, w2, w1f, w2f);
    s4_cvt  <<<(S4_B * S4_L * S4_DIN) / (256 * 8), 256, 0, stream>>>(data, d16);
    s4_kk   <<<(S4_H * (S4_L / 32) + 255) / 256, 256, 0, stream>>>(
        car, cai, ckr, cki, Dv, kkb, kks);
    s4_main <<<S4_B * 32, 256, 0, stream>>>(d16, b1, w1f, w2f, kkb, accb);
    s4_tail <<<S4_B, 256, 0, stream>>>(accb, b2, kks, Wg, bg, w3, b3, w4, b4, out);
}
